// InductiveActorNetwork_1915555414397
// MI455X (gfx1250) — matrix operand order repaired and checked by reading, not compiled, not run on hardware
//
#include <hip/hip_runtime.h>
#include <math.h>

// ---------------------------------------------------------------------------
// Problem constants (from reference)
// ---------------------------------------------------------------------------
#define NNODES 102400   // BS * BLK
#define NEDGES 409600
#define BSZ    4096
#define GDIM   64

#define ACT_NONE 0
#define ACT_RELU 1

typedef __attribute__((ext_vector_type(16))) _Float16 v16h;
typedef __attribute__((ext_vector_type(8)))  float    v8f;

// ---------------------------------------------------------------------------
// Utility kernels
// ---------------------------------------------------------------------------
__global__ void k_fill(float* __restrict__ p, float v, size_t n) {
  size_t i = (size_t)blockIdx.x * blockDim.x + threadIdx.x;
  if (i < n) p[i] = v;
}

// Pack fp32 weight W[K x N] (row-major) into WMMA B-fragment layout (f16):
// layout [K/32][Np/16][lane(32)][e(16)],  k = kt*32 + e + 16*(lane>>4),
// n = nt*16 + (lane&15).  Zero-pad n >= N.
__global__ void k_pack_b(const float* __restrict__ W, _Float16* __restrict__ dst,
                         int K, int N, int Np) {
  int idx = blockIdx.x * blockDim.x + threadIdx.x;
  if (idx >= K * Np) return;
  int e    = idx & 15;
  int lane = (idx >> 4) & 31;
  int tile = idx >> 9;
  int nT = Np >> 4;
  int kt = tile / nT, nt = tile - kt * nT;
  int k = kt * 32 + e + ((lane >> 4) << 4);
  int n = nt * 16 + (lane & 15);
  float v = (k < K && n < N) ? W[k * N + n] : 0.f;
  dst[idx] = (_Float16)v;
}

// ---------------------------------------------------------------------------
// Generic WMMA GEMM:  C[M,N] = act(A[M,K] @ B[K,N] + bias)
// A fp32 row-major, B pre-packed f16 fragments, C fp32 row-major (ld = N).
// One wave -> 16 rows x (NT*16) cols; the A fragment is loaded/converted once
// per K-step and reused across NT back-to-back WMMAs (A traffic / NT).
// 8 waves / block; K multiple of 32; M multiple of 128.
// ---------------------------------------------------------------------------
template <int NT>
__global__ __launch_bounds__(256) void k_wmma_gemm(
    const float* __restrict__ A, const _Float16* __restrict__ Bp,
    const float* __restrict__ bias, float* __restrict__ C,
    int M, int K, int N, int Np, int act)
{
  int lane = threadIdx.x & 31;
  int wave = threadIdx.x >> 5;
  int m0 = (blockIdx.x * 8 + wave) << 4;
  if (m0 >= M) return;                       // wave-uniform
  int l15 = lane & 15;
  int lhi = lane >> 4;
  int nT  = Np >> 4;
  int nt0 = blockIdx.y * NT;
  const float* arow = A + (size_t)(m0 + l15) * K;
  v8f zero = {};
  v8f acc[NT];
#pragma unroll
  for (int t = 0; t < NT; ++t) acc[t] = zero;

  for (int k0 = 0, kt = 0; k0 < K; k0 += 32, ++kt) {
    if (k0 + 32 < K) __builtin_prefetch(arow + k0 + 32, 0, 1);
    // A fragment: lane -> row (m0+l15); element e -> k = e + 16*lhi (matches packed B)
    v16h af;
#pragma unroll
    for (int e = 0; e < 16; ++e) {
      int k = k0 + e + (lhi << 4);
      af[e] = (_Float16)arow[k];
    }
    // NT B fragments: contiguous 32B per lane from packed buffer
    const _Float16* bb = Bp + (((size_t)kt * nT + nt0) * 32 + lane) * 16;
#pragma unroll
    for (int t = 0; t < NT; ++t) {
      v16h bf = *(const v16h*)(bb + (size_t)t * 512);
      acc[t] = __builtin_amdgcn_wmma_f32_16x16x32_f16(false, af, false, bf,
                                                      (short)0, acc[t], false, false);
    }
  }
#pragma unroll
  for (int t = 0; t < NT; ++t) {
    int n = ((nt0 + t) << 4) + l15;
    if (n < N) {
      float bv = bias ? bias[n] : 0.f;
#pragma unroll
      for (int r = 0; r < 8; ++r) {
        int m = m0 + r + (lhi << 3);         // C/D layout: vgpr r, lane-half
        float v = acc[t][r] + bv;
        if (act == ACT_RELU) v = fmaxf(v, 0.f);
        C[(size_t)m * N + n] = v;
      }
    }
  }
}

// ---------------------------------------------------------------------------
// GCN pieces
// ---------------------------------------------------------------------------
__global__ void k_deg(const int* __restrict__ dst, float* __restrict__ degf, int E) {
  int e = blockIdx.x * blockDim.x + threadIdx.x;
  if (e < E) atomicAdd(&degf[dst[e]], 1.f);
}

__global__ void k_rsqrt(const float* __restrict__ degf, float* __restrict__ dis, int n) {
  int i = blockIdx.x * blockDim.x + threadIdx.x;
  if (i < n) dis[i] = rsqrtf(degf[i]);
}

// One thread handles 4 channels of one edge message (b128 load + 4 atomics)
__global__ void k_gcn_scatter4(const float* __restrict__ h, const int* __restrict__ src,
                               const int* __restrict__ dst, const float* __restrict__ dis,
                               float* __restrict__ accum, int Hc, int logH4)
{
  size_t idx = (size_t)blockIdx.x * blockDim.x + threadIdx.x;
  size_t e = idx >> logH4;
  if (e >= (size_t)NEDGES) return;
  int c4 = (int)(idx & (size_t)((Hc >> 2) - 1)) << 2;
  int s = src[e], d = dst[e];
  float w = dis[s] * dis[d];
  const float4 hv = *(const float4*)(h + (size_t)s * Hc + c4);
  float* ap = accum + (size_t)d * Hc + c4;
  atomicAdd(ap + 0, hv.x * w);
  atomicAdd(ap + 1, hv.y * w);
  atomicAdd(ap + 2, hv.z * w);
  atomicAdd(ap + 3, hv.w * w);
}

// accum = relu(accum + h * dis^2 + bias)  (self-loop + bias + relu), in place
__global__ void k_gcn_epi(float* __restrict__ accum, const float* __restrict__ h,
                          const float* __restrict__ dis, const float* __restrict__ bias,
                          int Hc, int logH, size_t total)
{
  size_t idx = (size_t)blockIdx.x * blockDim.x + threadIdx.x;
  if (idx >= total) return;
  size_t nid = idx >> logH;
  int c = (int)(idx & (size_t)(Hc - 1));
  float di = dis[nid];
  float v = accum[idx] + h[idx] * di * di + bias[c];
  accum[idx] = fmaxf(v, 0.f);
}

// ---------------------------------------------------------------------------
// Global vector / attention pooling
// ---------------------------------------------------------------------------
__global__ void k_ginit(const float* __restrict__ gv, const float* __restrict__ W,
                        const float* __restrict__ b, float* __restrict__ g)
{
  int idx = blockIdx.x * blockDim.x + threadIdx.x;   // BSZ*64
  int bb = idx >> 6, c = idx & 63;
  float s = b[c];
#pragma unroll
  for (int k = 0; k < 3; ++k) s += gv[bb * 3 + k] * W[k * 64 + c];
  g[idx] = s;
}

// Build V[BSZ*25, dim]: slots 0..5 servers (masked), 6..15 users (masked), 16..24 routers
__global__ void k_gather_v(const float* __restrict__ xc, int dim,
                           const int* __restrict__ srv, const int* __restrict__ usr,
                           const int* __restrict__ rtr,
                           const float* __restrict__ s_mask, const float* __restrict__ u_mask,
                           float* __restrict__ V, size_t total)
{
  size_t idx = (size_t)blockIdx.x * blockDim.x + threadIdx.x;
  if (idx >= total) return;
  int c = (int)(idx % (size_t)dim);
  size_t row = idx / (size_t)dim;
  int slot = (int)(row % 25);
  int b = (int)(row / 25);
  int src; float m;
  if (slot < 6)       { src = srv[b * 6 + slot];       m = s_mask[b * 6 + slot]; }
  else if (slot < 16) { src = usr[b * 10 + slot - 6];  m = u_mask[b * 10 + slot - 6]; }
  else                { src = rtr[b * 9 + slot - 16];  m = 1.f; }
  V[idx] = xc[(size_t)src * dim + c] * m;
}

// Per subnet: softmax over feature dim of attL rows, masked weighted sum of featL
__global__ __launch_bounds__(256) void k_pool_reduce(
    const float* __restrict__ attL, const float* __restrict__ featL,
    const float* __restrict__ s_mask, const float* __restrict__ u_mask,
    float* __restrict__ pooled, int h)
{
  __shared__ float smax[25], ssum[25];
  int b = blockIdx.x;
  int lane = threadIdx.x & 31, wave = threadIdx.x >> 5;
  for (int slot = wave; slot < 25; slot += 8) {
    const float* row = attL + ((size_t)b * 25 + slot) * h;
    float mx = -1e30f;
    for (int c = lane; c < h; c += 32) mx = fmaxf(mx, row[c]);
    for (int o = 16; o; o >>= 1) mx = fmaxf(mx, __shfl_xor(mx, o, 32));
    float sm = 0.f;
    for (int c = lane; c < h; c += 32) sm += __expf(row[c] - mx);
    for (int o = 16; o; o >>= 1) sm += __shfl_xor(sm, o, 32);
    if (lane == 0) { smax[slot] = mx; ssum[slot] = sm; }
  }
  __syncthreads();
  for (int c = threadIdx.x; c < h; c += 256) {
    float acc = 0.f;
    for (int slot = 0; slot < 25; ++slot) {
      float m = (slot < 6) ? s_mask[b * 6 + slot]
              : (slot < 16) ? u_mask[b * 10 + slot - 6] : 1.f;
      size_t r = ((size_t)b * 25 + slot) * h + c;
      acc += (__expf(attL[r] - smax[slot]) / ssum[slot]) * featL[r] * m;
    }
    pooled[(size_t)b * h + c] = acc;
  }
}

__global__ void k_concat_pg(const float* __restrict__ pooled, const float* __restrict__ g,
                            float* __restrict__ gcat, int h, size_t total)
{
  size_t idx = (size_t)blockIdx.x * blockDim.x + threadIdx.x;
  if (idx >= total) return;
  int w = h + GDIM;
  int c = (int)(idx % (size_t)w);
  int b = (int)(idx / (size_t)w);
  gcat[idx] = (c < h) ? pooled[(size_t)b * h + c] : g[b * GDIM + (c - h)];
}

__global__ void k_g_update(float* __restrict__ g, const float* __restrict__ t, int n) {
  int i = blockIdx.x * blockDim.x + threadIdx.x;
  if (i < n) g[i] += tanhf(t[i]);
}

// ---------------------------------------------------------------------------
// Heads
// ---------------------------------------------------------------------------
// A_node[BSZ*16, 128]: cols 0..63 masked host features, 64..127 broadcast g
__global__ void k_gather_node(const float* __restrict__ x2,
                              const int* __restrict__ srv, const int* __restrict__ usr,
                              const float* __restrict__ s_mask, const float* __restrict__ u_mask,
                              const float* __restrict__ g, float* __restrict__ out)
{
  size_t idx = (size_t)blockIdx.x * blockDim.x + threadIdx.x;  // BSZ*16*128
  int c = (int)(idx & 127);
  size_t row = idx >> 7;
  int slot = (int)(row & 15);
  int b = (int)(row >> 4);
  float v;
  if (c < 64) {
    int sidx; float m;
    if (slot < 6) { sidx = srv[b * 6 + slot];      m = s_mask[b * 6 + slot]; }
    else          { sidx = usr[b * 10 + slot - 6]; m = u_mask[b * 10 + slot - 6]; }
    v = x2[(size_t)sidx * 64 + c] * m;
  } else {
    v = g[b * GDIM + (c - 64)];
  }
  out[idx] = v;
}

// stage[b*81 + a*16 + slot] = nout[(b*16+slot)*4 + a] * mask
__global__ void k_node_fin(const float* __restrict__ nout,
                           const float* __restrict__ s_mask, const float* __restrict__ u_mask,
                           float* __restrict__ stage)
{
  int idx = blockIdx.x * blockDim.x + threadIdx.x;  // BSZ*64
  int b = idx >> 6, r = idx & 63;
  int a = r >> 4, slot = r & 15;
  float m = (slot < 6) ? s_mask[b * 6 + slot] : u_mask[b * 10 + slot - 6];
  stage[b * 81 + a * 16 + slot] = nout[(size_t)(b * 16 + slot) * 4 + a] * m;
}

__global__ void k_gather_rows(const float* __restrict__ src, const int* __restrict__ rows,
                              float* __restrict__ out, size_t total)
{
  size_t idx = (size_t)blockIdx.x * blockDim.x + threadIdx.x;  // rows*64
  if (idx >= total) return;
  int c = (int)(idx & 63);
  size_t r = idx >> 6;
  out[idx] = src[(size_t)rows[r] * 64 + c];
}

// eacat[32768, 96]: c<32 -> es2*ed2 ; else g repeated per 8 edges
__global__ void k_edge_cat(const float* __restrict__ es2, const float* __restrict__ ed2,
                           const float* __restrict__ g, float* __restrict__ out, size_t total)
{
  size_t idx = (size_t)blockIdx.x * blockDim.x + threadIdx.x;
  if (idx >= total) return;
  int c = (int)(idx % 96);
  size_t r = idx / 96;
  float v = (c < 32) ? es2[r * 32 + c] * ed2[r * 32 + c]
                     : g[(r >> 3) * GDIM + (c - 32)];
  out[idx] = v;
}

// stage[b*81 + 64 + a*8 + e] = eaout[(b*8+e)*2 + a]
__global__ void k_edge_fin(const float* __restrict__ eaout, float* __restrict__ stage)
{
  int idx = blockIdx.x * blockDim.x + threadIdx.x;  // BSZ*16
  int b = idx >> 4, r = idx & 15;
  int a = r >> 3, e = r & 7;
  stage[b * 81 + 64 + a * 8 + e] = eaout[(size_t)(b * 8 + e) * 2 + a];
}

__global__ void k_glb_fin(const float* __restrict__ gout, float* __restrict__ stage, int n)
{
  int b = blockIdx.x * blockDim.x + threadIdx.x;
  if (b < n) stage[b * 81 + 80] = gout[b];
}

// masked softmax over 81 logits: 0 -> -inf
__global__ __launch_bounds__(128) void k_final_softmax(const float* __restrict__ st,
                                                       float* __restrict__ out)
{
  __shared__ float sh[128];
  int b = blockIdx.x, t = threadIdx.x;
  float v = -INFINITY;
  if (t < 81) { v = st[b * 81 + t]; if (v == 0.f) v = -INFINITY; }
  sh[t] = v; __syncthreads();
  for (int o = 64; o; o >>= 1) { if (t < o) sh[t] = fmaxf(sh[t], sh[t + o]); __syncthreads(); }
  float mx = sh[0]; __syncthreads();
  float ev = (t < 81) ? __expf(v - mx) : 0.f;
  sh[t] = ev; __syncthreads();
  for (int o = 64; o; o >>= 1) { if (t < o) sh[t] += sh[t + o]; __syncthreads(); }
  float sum = sh[0];
  if (t < 81) out[b * 81 + t] = ev / sum;
}

// ---------------------------------------------------------------------------
// Host orchestration
// ---------------------------------------------------------------------------
static inline unsigned cdiv(size_t a, size_t b) { return (unsigned)((a + b - 1) / b); }

extern "C" void kernel_launch(void* const* d_in, const int* in_sizes, int n_in,
                              void* d_out, int out_size, void* d_ws, size_t ws_size,
                              hipStream_t stream) {
  (void)in_sizes; (void)n_in; (void)out_size; (void)ws_size;
  // --- input mapping: top-level dict in insertion order, params pytree with
  //     sorted dict keys (jax tree order), {b,w} per linear, lists in order ---
  const float* x        = (const float*)d_in[0];
  const float* gv       = (const float*)d_in[1];
  const float* s_mask   = (const float*)d_in[2];
  const float* u_mask   = (const float*)d_in[3];
  const float* conv1_b  = (const float*)d_in[4];
  const float* conv1_w  = (const float*)d_in[5];
  const float* conv2_b  = (const float*)d_in[6];
  const float* conv2_w  = (const float*)d_in[7];
  const float* edge0_b  = (const float*)d_in[8];
  const float* edge0_w  = (const float*)d_in[9];
  const float* edge1_b  = (const float*)d_in[10];
  const float* edge1_w  = (const float*)d_in[11];
  const float* eout_b   = (const float*)d_in[12];
  const float* eout_w   = (const float*)d_in[13];
  const float* g0a_b    = (const float*)d_in[14];
  const float* g0a_w    = (const float*)d_in[15];
  const float* g0f_b    = (const float*)d_in[16];
  const float* g0f_w    = (const float*)d_in[17];
  const float* g0g_b    = (const float*)d_in[18];
  const float* g0g_w    = (const float*)d_in[19];
  const float* g1a_b    = (const float*)d_in[20];
  const float* g1a_w    = (const float*)d_in[21];
  const float* g1f_b    = (const float*)d_in[22];
  const float* g1f_w    = (const float*)d_in[23];
  const float* g1g_b    = (const float*)d_in[24];
  const float* g1g_w    = (const float*)d_in[25];
  const float* g2a_b    = (const float*)d_in[26];
  const float* g2a_w    = (const float*)d_in[27];
  const float* g2f_b    = (const float*)d_in[28];
  const float* g2f_w    = (const float*)d_in[29];
  const float* g2g_b    = (const float*)d_in[30];
  const float* g2g_w    = (const float*)d_in[31];
  const float* glb0_b   = (const float*)d_in[32];
  const float* glb0_w   = (const float*)d_in[33];
  const float* glb1_b   = (const float*)d_in[34];
  const float* glb1_w   = (const float*)d_in[35];
  const float* gn_b     = (const float*)d_in[36];
  const float* gn_w     = (const float*)d_in[37];
  const float* node0_b  = (const float*)d_in[38];
  const float* node0_w  = (const float*)d_in[39];
  const float* node1_b  = (const float*)d_in[40];
  const float* node1_w  = (const float*)d_in[41];
  const float* node2_b  = (const float*)d_in[42];
  const float* node2_w  = (const float*)d_in[43];
  const int*   ei       = (const int*)d_in[44];
  const int*   srv_idx  = (const int*)d_in[45];
  const int*   usr_idx  = (const int*)d_in[46];
  const int*   rtrs     = (const int*)d_in[47];
  const int*   ae_src   = (const int*)d_in[48];
  const int*   ae_dst   = (const int*)d_in[49];
  const int* esrc = ei;
  const int* edst = ei + NEDGES;

  // --- workspace layout ---
  char* ws = (char*)d_ws;
  size_t off = 0;
  auto alloc = [&](size_t bytes) -> void* {
    void* p = ws + off; off = (off + bytes + 255) & ~(size_t)255; return p;
  };
  _Float16* packBase = (_Float16*)alloc(1 << 20);                 // all packed weights
  float* degf   = (float*)alloc((size_t)NNODES * 4);
  float* dis    = (float*)alloc((size_t)NNODES * 4);
  float* g      = (float*)alloc((size_t)BSZ * GDIM * 4);
  float* pooled = (float*)alloc((size_t)BSZ * 256 * 4);
  float* gcat   = (float*)alloc((size_t)BSZ * 320 * 4);
  float* gtmp   = (float*)alloc((size_t)BSZ * GDIM * 4);
  float* stage  = (float*)alloc((size_t)BSZ * 81 * 4);
  float* bufH1  = (float*)alloc((size_t)NNODES * 256 * 4);        // h1; reused as V
  float* x1     = (float*)alloc((size_t)NNODES * 256 * 4);        // acc1 -> x1
  float* attL   = (float*)alloc((size_t)NNODES * 256 * 4);        // reused: node head
  float* featL  = (float*)alloc((size_t)NNODES * 256 * 4);        // reused: edge head
  float* h2     = (float*)alloc((size_t)NNODES * 64 * 4);
  float* x2     = (float*)alloc((size_t)NNODES * 64 * 4);         // acc2 -> x2
  // sub-regions inside attL (used only after pools are done)
  float* A_node = attL;
  float* nh1    = attL + (size_t)65536 * 128;
  float* nh2    = nh1  + (size_t)65536 * 64;
  float* nout   = nh2  + (size_t)65536 * 32;
  // sub-regions inside featL (used only after pools are done)
  float* esin  = featL;
  float* edin  = esin  + (size_t)32768 * 64;
  float* es1   = edin  + (size_t)32768 * 64;
  float* ed1   = es1   + (size_t)32768 * 64;
  float* es2   = ed1   + (size_t)32768 * 64;
  float* ed2   = es2   + (size_t)32768 * 32;
  float* eacat = ed2   + (size_t)32768 * 32;
  float* eaout = eacat + (size_t)32768 * 96;
  float* ghid  = eaout + (size_t)32768 * 2;
  float* gout  = ghid  + (size_t)BSZ * 32;

  // --- weight packing (every call; deterministic) ---
  size_t phalf = 0;
  auto packW = [&](const float* W, int K, int N) -> const _Float16* {
    int Np = (N + 15) & ~15;
    _Float16* dst = packBase + phalf;
    phalf += (size_t)K * Np;
    k_pack_b<<<cdiv((size_t)K * Np, 256), 256, 0, stream>>>(W, dst, K, N, Np);
    return dst;
  };
  const _Float16* pc1 = packW(conv1_w, 64, 256);
  const _Float16* pc2 = packW(conv2_w, 256, 64);
  const _Float16* p0a = packW(g0a_w, 64, 256);
  const _Float16* p0f = packW(g0f_w, 64, 256);
  const _Float16* p0g = packW(g0g_w, 320, 64);
  const _Float16* p1a = packW(g1a_w, 256, 256);
  const _Float16* p1f = packW(g1f_w, 256, 256);
  const _Float16* p1g = packW(g1g_w, 320, 64);
  const _Float16* p2a = packW(g2a_w, 64, 64);
  const _Float16* p2f = packW(g2f_w, 64, 64);
  const _Float16* p2g = packW(g2g_w, 128, 64);
  const _Float16* pn0 = packW(node0_w, 128, 64);
  const _Float16* pn1 = packW(node1_w, 64, 32);
  const _Float16* pn2 = packW(node2_w, 32, 4);
  const _Float16* pe0 = packW(edge0_w, 64, 64);
  const _Float16* pe1 = packW(edge1_w, 64, 32);
  const _Float16* peo = packW(eout_w, 96, 2);
  const _Float16* pg0 = packW(glb0_w, 64, 32);
  const _Float16* pg1 = packW(glb1_w, 32, 1);

  // NT = widest of {8,4,2,1} dividing the n-tile count (maximizes A reuse)
  auto gemm = [&](const float* A, const _Float16* Bp, const float* bias, float* C,
                  int M, int K, int N, int act) {
    int Np = (N + 15) & ~15;
    int nTiles = Np >> 4;
    int NT = (nTiles % 8 == 0) ? 8 : (nTiles % 4 == 0) ? 4 : (nTiles % 2 == 0) ? 2 : 1;
    dim3 grid(M / 128, nTiles / NT);
    switch (NT) {
      case 8: k_wmma_gemm<8><<<grid, 256, 0, stream>>>(A, Bp, bias, C, M, K, N, Np, act); break;
      case 4: k_wmma_gemm<4><<<grid, 256, 0, stream>>>(A, Bp, bias, C, M, K, N, Np, act); break;
      case 2: k_wmma_gemm<2><<<grid, 256, 0, stream>>>(A, Bp, bias, C, M, K, N, Np, act); break;
      default: k_wmma_gemm<1><<<grid, 256, 0, stream>>>(A, Bp, bias, C, M, K, N, Np, act); break;
    }
  };

  // --- degrees: deg = 1 (self loop) + indeg; dis = rsqrt(deg) ---
  k_fill<<<cdiv(NNODES, 256), 256, 0, stream>>>(degf, 1.f, NNODES);
  k_deg<<<cdiv(NEDGES, 256), 256, 0, stream>>>(edst, degf, NEDGES);
  k_rsqrt<<<cdiv(NNODES, 256), 256, 0, stream>>>(degf, dis, NNODES);

  // --- GCN1: x1 = relu(scatter(x @ W1) + selfloop + b) ---
  gemm(x, pc1, nullptr, bufH1, NNODES, 64, 256, ACT_NONE);
  k_fill<<<cdiv((size_t)NNODES * 256, 256), 256, 0, stream>>>(x1, 0.f, (size_t)NNODES * 256);
  k_gcn_scatter4<<<cdiv((size_t)NEDGES * 64, 256), 256, 0, stream>>>(bufH1, esrc, edst, dis, x1, 256, 6);
  k_gcn_epi<<<cdiv((size_t)NNODES * 256, 256), 256, 0, stream>>>(x1, bufH1, dis, conv1_b, 256, 8, (size_t)NNODES * 256);

  // --- GCN2: x2 = relu(scatter(x1 @ W2) + selfloop + b) ---
  gemm(x1, pc2, nullptr, h2, NNODES, 256, 64, ACT_NONE);
  k_fill<<<cdiv((size_t)NNODES * 64, 256), 256, 0, stream>>>(x2, 0.f, (size_t)NNODES * 64);
  k_gcn_scatter4<<<cdiv((size_t)NEDGES * 16, 256), 256, 0, stream>>>(h2, esrc, edst, dis, x2, 64, 4);
  k_gcn_epi<<<cdiv((size_t)NNODES * 64, 256), 256, 0, stream>>>(x2, h2, dis, conv2_b, 64, 6, (size_t)NNODES * 64);

  // --- g init: g = gv @ W[3,64] + b ---
  k_ginit<<<cdiv((size_t)BSZ * GDIM, 256), 256, 0, stream>>>(gv, gn_w, gn_b, g);

  float* V = bufH1;  // h1 region is free now
  auto pool = [&](const float* xc, int dim, int h,
                  const _Float16* pa, const float* ba,
                  const _Float16* pf, const float* bf,
                  const _Float16* pg_, const float* bg) {
    size_t vtotal = (size_t)BSZ * 25 * dim;
    k_gather_v<<<cdiv(vtotal, 256), 256, 0, stream>>>(xc, dim, srv_idx, usr_idx, rtrs,
                                                      s_mask, u_mask, V, vtotal);
    gemm(V, pa, ba, attL,  BSZ * 25, dim, h, ACT_NONE);
    gemm(V, pf, bf, featL, BSZ * 25, dim, h, ACT_NONE);
    k_pool_reduce<<<BSZ, 256, 0, stream>>>(attL, featL, s_mask, u_mask, pooled, h);
    size_t ctotal = (size_t)BSZ * (h + GDIM);
    k_concat_pg<<<cdiv(ctotal, 256), 256, 0, stream>>>(pooled, g, gcat, h, ctotal);
    gemm(gcat, pg_, bg, gtmp, BSZ, h + GDIM, GDIM, ACT_NONE);
    k_g_update<<<cdiv((size_t)BSZ * GDIM, 256), 256, 0, stream>>>(g, gtmp, BSZ * GDIM);
  };
  pool(x,  64,  256, p0a, g0a_b, p0f, g0f_b, p0g, g0g_b);   // g0
  pool(x1, 256, 256, p1a, g1a_b, p1f, g1f_b, p1g, g1g_b);   // g1
  pool(x2, 64,  64,  p2a, g2a_b, p2f, g2f_b, p2g, g2g_b);   // g2

  // --- node head ---
  k_gather_node<<<cdiv((size_t)BSZ * 16 * 128, 256), 256, 0, stream>>>(
      x2, srv_idx, usr_idx, s_mask, u_mask, g, A_node);
  gemm(A_node, pn0, node0_b, nh1,  BSZ * 16, 128, 64, ACT_RELU);
  gemm(nh1,    pn1, node1_b, nh2,  BSZ * 16, 64,  32, ACT_RELU);
  gemm(nh2,    pn2, node2_b, nout, BSZ * 16, 32,  4,  ACT_NONE);
  k_node_fin<<<cdiv((size_t)BSZ * 64, 256), 256, 0, stream>>>(nout, s_mask, u_mask, stage);

  // --- edge head ---
  k_gather_rows<<<cdiv((size_t)32768 * 64, 256), 256, 0, stream>>>(x2, ae_src, esin, (size_t)32768 * 64);
  k_gather_rows<<<cdiv((size_t)32768 * 64, 256), 256, 0, stream>>>(x2, ae_dst, edin, (size_t)32768 * 64);
  gemm(esin, pe0, edge0_b, es1, 32768, 64, 64, ACT_RELU);
  gemm(es1,  pe1, edge1_b, es2, 32768, 64, 32, ACT_RELU);
  gemm(edin, pe0, edge0_b, ed1, 32768, 64, 64, ACT_RELU);
  gemm(ed1,  pe1, edge1_b, ed2, 32768, 64, 32, ACT_RELU);
  k_edge_cat<<<cdiv((size_t)32768 * 96, 256), 256, 0, stream>>>(es2, ed2, g, eacat, (size_t)32768 * 96);
  gemm(eacat, peo, eout_b, eaout, 32768, 96, 2, ACT_NONE);
  k_edge_fin<<<cdiv((size_t)BSZ * 16, 256), 256, 0, stream>>>(eaout, stage);

  // --- global head ---
  gemm(g,    pg0, glb0_b, ghid, BSZ, 64, 32, ACT_RELU);
  gemm(ghid, pg1, glb1_b, gout, BSZ, 32, 1,  ACT_NONE);
  k_glb_fin<<<cdiv(BSZ, 256), 256, 0, stream>>>(gout, stage, BSZ);

  // --- final masked softmax -> d_out [BSZ, 81] ---
  k_final_softmax<<<BSZ, 128, 0, stream>>>(stage, (float*)d_out);
}
